// InfiniAssociativeMemoryCell_52201032516255
// MI455X (gfx1250) — matrix operand order repaired and checked by reading, not compiled, not run on hardware
//
#include <hip/hip_runtime.h>
#include <math.h>

// ---------------- problem constants ----------------
static const int NB   = 4;     // batch
static const int SEQ  = 4224;  // sequence length
static const int DIM  = 1024;  // model dim
static const int DMQ  = 64;    // pre-dpfp key dim
static const int DK   = 128;   // dpfp key dim
static const int SEGL = 528;   // segment length (512 + 16)
static const int NSEG = 8;
static const int MT   = 16;    // memory tokens
#define EPSF 1e-5f

typedef __attribute__((ext_vector_type(16))) __bf16 v16bf;
typedef __attribute__((ext_vector_type(8)))  float  v8f;
typedef unsigned short ushort_t;

union BF16x16 {
  v16bf    bf;
  ushort_t u[16];
  uint4    q[2];
};

// Native f32 -> bf16 convert: fptrunc lowers to v_cvt_pk_bf16_f32 on gfx1250,
// replacing the 3-VALU manual round-to-nearest-even sequence.
static __device__ inline ushort_t f2bf(float f) {
  return __builtin_bit_cast(ushort_t, static_cast<__bf16>(f));
}
static __device__ inline float bf2f(ushort_t h) {
  unsigned u = ((unsigned)h) << 16;
  return __builtin_bit_cast(float, u);
}
// Unpack two bf16 from one packed dword (low/high half).
static __device__ inline float bflo(unsigned u) {
  return __builtin_bit_cast(float, u << 16);
}
static __device__ inline float bfhi(unsigned u) {
  return __builtin_bit_cast(float, u & 0xffff0000u);
}
static __device__ inline v8f wmma_bf16(v16bf a, v16bf b, v8f c) {
  // D = A(16x32 bf16) * B(32x16 bf16) + C(16x16 f32)
  return __builtin_amdgcn_wmma_f32_16x16x32_bf16(false, a, false, b, (short)0, c,
                                                 false, false);
}

// ---------------------------------------------------------------------------
// Pre-swizzle a KxN f32 weight into bf16 WMMA B-panels.
// Panel (ktile,ntile) covers K[ktile*32 .. +32) x N[ntile*16 .. +16).
// Lane l holds N = ntile*16 + l%16, element i holds K = ktile*32 + 8*(l/16) + i (i<8), ktile*32 + 16 + 8*(l/16) + (i-8) (i>=8).
// ---------------------------------------------------------------------------
__global__ void kprep(const float* __restrict__ W, ushort_t* __restrict__ pan,
                      int K, int N) {
  int tid = blockIdx.x * blockDim.x + threadIdx.x;
  int NT = N >> 4;
  int total = (K >> 5) * NT * 32;
  if (tid >= total) return;
  int l = tid & 31, p = tid >> 5;
  int ktile = p / NT, ntile = p - ktile * NT;
  int g = l >> 4, m = l & 15;
  ushort_t* o = pan + (size_t)tid * 16;
  int nn = ntile * 16 + m;
#pragma unroll
  for (int i = 0; i < 16; ++i) {
    int kk = ktile * 32 + ((i < 8) ? (8 * g + i) : (16 + 8 * g + (i - 8)));
    o[i] = f2bf(W[(size_t)kk * N + nn]);
  }
}

__global__ void kzero(unsigned* __restrict__ p, long n) {
  long i = (long)blockIdx.x * blockDim.x + threadIdx.x;
  if (i < n) p[i] = 0u;
}

// ---------------------------------------------------------------------------
// Q = dpfp(hidden @ W_mq)  ->  bf16 [NB*SEQ, 128]
// One wave per 16-row tile; 32 K-steps x 4 N-tiles of WMMA; dpfp via LDS.
// ---------------------------------------------------------------------------
__global__ __launch_bounds__(32) void kq(const float* __restrict__ hid,
                                         const ushort_t* __restrict__ panQ,
                                         ushort_t* __restrict__ Qbf) {
  __shared__ float y[16 * 64];
  int l = threadIdx.x, m = l & 15, g = l >> 4;
  int rowbase = blockIdx.x * 16;
  const float* arow = hid + (size_t)(rowbase + m) * DIM;
  v8f acc[4] = {};
  for (int kt = 0; kt < 32; ++kt) {
    const float* a0 = arow + kt * 32 + 8 * g;
    float t[16];
    *(float4*)(t + 0)  = *(const float4*)(a0);
    *(float4*)(t + 4)  = *(const float4*)(a0 + 4);
    *(float4*)(t + 8)  = *(const float4*)(a0 + 16);
    *(float4*)(t + 12) = *(const float4*)(a0 + 20);
    __builtin_prefetch(a0 + 32, 0, 1);  // global_prefetch next K-step
    BF16x16 A;
#pragma unroll
    for (int i = 0; i < 16; ++i) A.u[i] = f2bf(t[i]);
#pragma unroll
    for (int nt = 0; nt < 4; ++nt) {
      BF16x16 Bv;
      const uint4* bp = (const uint4*)(panQ + ((size_t)(kt * 4 + nt) * 32 + l) * 16);
      Bv.q[0] = bp[0]; Bv.q[1] = bp[1];
      acc[nt] = wmma_bf16(A.bf, Bv.bf, acc[nt]);
    }
  }
#pragma unroll
  for (int nt = 0; nt < 4; ++nt)
#pragma unroll
    for (int r = 0; r < 8; ++r)
      y[(r + 8 * g) * 64 + nt * 16 + m] = acc[nt][r];
  __syncthreads();
  for (int idx = l; idx < 16 * DK; idx += 32) {
    int row = idx >> 7, i = idx & 127;
    int ip = (i + 127) & 127;
    float yi = (i  < 64) ? y[row * 64 + i]  : -y[row * 64 + (i - 64)];
    float yp = (ip < 64) ? y[row * 64 + ip] : -y[row * 64 + (ip - 64)];
    float q = fmaxf(yi, 0.f) * fmaxf(yp, 0.f);
    Qbf[(size_t)(rowbase + row) * DK + i] = f2bf(q);
  }
}

// ---------------------------------------------------------------------------
// assoc = (Q_seg @ W_mem) / (Q_seg . z + eps);  out = mo + tanh(assoc);
// stage mem tokens (assoc + mo) for rows >= 512.
// Grid: x = b*33 + rowtile, y = 64-col group. One wave, 16 WMMAs (K=128).
// ---------------------------------------------------------------------------
__global__ __launch_bounds__(32) void kassoc(const ushort_t* __restrict__ Qbf,
                                             const ushort_t* __restrict__ panM,
                                             const float* __restrict__ z,
                                             const float* __restrict__ mo,
                                             float* __restrict__ out,
                                             float* __restrict__ memtok,
                                             int s) {
  __shared__ float den[16];
  int l = threadIdx.x, m = l & 15, g = l >> 4;
  int b = blockIdx.x / 33, rt = blockIdx.x % 33, ng = blockIdx.y;
  int lrow0 = rt * 16;
  size_t qrow0 = (size_t)b * SEQ + (size_t)s * SEGL + lrow0;
  if (l < 16) {
    const uint4*  q4 = (const uint4*)(Qbf + (qrow0 + l) * DK);
    const float4* z4 = (const float4*)(z + b * DK);
    float d = EPSF;
#pragma unroll 4
    for (int c = 0; c < 16; ++c) {   // 8 bf16 + 8 f32 per iter, b128 loads
      uint4  qv = q4[c];
      float4 z0 = z4[2 * c], z1 = z4[2 * c + 1];
      d += bflo(qv.x) * z0.x + bfhi(qv.x) * z0.y +
           bflo(qv.y) * z0.z + bfhi(qv.y) * z0.w +
           bflo(qv.z) * z1.x + bfhi(qv.z) * z1.y +
           bflo(qv.w) * z1.z + bfhi(qv.w) * z1.w;
    }
    den[l] = d;
  }
  __syncthreads();
  v8f acc[4] = {};
  const ushort_t* arow = Qbf + (qrow0 + m) * DK;
  const ushort_t* pb = panM + (size_t)b * (4 * 64 * 32 * 16);
  for (int kt = 0; kt < 4; ++kt) {
    BF16x16 A;
    const ushort_t* a0 = arow + kt * 32 + 8 * g;
    A.q[0] = *(const uint4*)a0;
    A.q[1] = *(const uint4*)(a0 + 16);
#pragma unroll
    for (int nt = 0; nt < 4; ++nt) {
      BF16x16 Bv;
      const uint4* bp =
          (const uint4*)(pb + ((size_t)(kt * 64 + ng * 4 + nt) * 32 + l) * 16);
      Bv.q[0] = bp[0]; Bv.q[1] = bp[1];
      acc[nt] = wmma_bf16(A.bf, Bv.bf, acc[nt]);
    }
  }
#pragma unroll
  for (int nt = 0; nt < 4; ++nt)
#pragma unroll
    for (int r = 0; r < 8; ++r) {
      int lr = lrow0 + r + 8 * g;
      int c = ng * 64 + nt * 16 + m;
      size_t oi = ((size_t)b * SEQ + (size_t)s * SEGL + lr) * DIM + c;
      float a = acc[nt][r] / den[r + 8 * g];
      float mval = mo[oi];
      out[oi] = mval + tanhf(a);
      if (lr >= SEGL - MT)
        memtok[((size_t)b * MT + (lr - (SEGL - MT))) * DIM + c] = a + mval;
    }
}

// ---------------------------------------------------------------------------
// k = dpfp(mem @ W_mk); den_m = k.z_old + eps; z += sum_t k.
// Grid: NB blocks x 128 threads (4 waves, one N-tile each).
// ---------------------------------------------------------------------------
__global__ __launch_bounds__(128) void kupd_a(const float* __restrict__ memtok,
                                              const ushort_t* __restrict__ panK,
                                              float* __restrict__ z,
                                              float* __restrict__ kf32,
                                              ushort_t* __restrict__ kbf,
                                              float* __restrict__ denm) {
  __shared__ float y[16 * 64];
  __shared__ float kl[16 * DK];
  int b = blockIdx.x;
  int tid = threadIdx.x, w = tid >> 5, l = tid & 31, m = l & 15, g = l >> 4;
  const float* arow = memtok + ((size_t)b * MT + m) * DIM;
  v8f acc = {};
  for (int kt = 0; kt < 32; ++kt) {
    const float* a0 = arow + kt * 32 + 8 * g;
    float t[16];
    *(float4*)(t + 0)  = *(const float4*)(a0);
    *(float4*)(t + 4)  = *(const float4*)(a0 + 4);
    *(float4*)(t + 8)  = *(const float4*)(a0 + 16);
    *(float4*)(t + 12) = *(const float4*)(a0 + 20);
    BF16x16 A;
#pragma unroll
    for (int i = 0; i < 16; ++i) A.u[i] = f2bf(t[i]);
    BF16x16 Bv;
    const uint4* bp = (const uint4*)(panK + ((size_t)(kt * 4 + w) * 32 + l) * 16);
    Bv.q[0] = bp[0]; Bv.q[1] = bp[1];
    acc = wmma_bf16(A.bf, Bv.bf, acc);
  }
#pragma unroll
  for (int r = 0; r < 8; ++r) y[(r + 8 * g) * 64 + w * 16 + m] = acc[r];
  __syncthreads();
  for (int idx = tid; idx < MT * DK; idx += 128) {
    int row = idx >> 7, i = idx & 127, ip = (i + 127) & 127;
    float yi = (i  < 64) ? y[row * 64 + i]  : -y[row * 64 + (i - 64)];
    float yp = (ip < 64) ? y[row * 64 + ip] : -y[row * 64 + (ip - 64)];
    float q = fmaxf(yi, 0.f) * fmaxf(yp, 0.f);
    kl[idx] = q;
    kf32[(size_t)b * MT * DK + idx] = q;
    kbf[(size_t)b * MT * DK + idx] = f2bf(q);
  }
  __syncthreads();
  if (tid < MT) {
    const float4* z4 = (const float4*)(z + b * DK);
    float d = EPSF;
#pragma unroll 4
    for (int c = 0; c < 32; ++c) {
      float4 zz = z4[c];
      d += kl[tid * DK + 4 * c + 0] * zz.x + kl[tid * DK + 4 * c + 1] * zz.y +
           kl[tid * DK + 4 * c + 2] * zz.z + kl[tid * DK + 4 * c + 3] * zz.w;
    }
    denm[b * MT + tid] = d;
  }
  __syncthreads();  // all old-z reads done before update
  if (tid < DK) {
    float sum = 0.f;
    for (int t2 = 0; t2 < MT; ++t2) sum += kl[t2 * DK + tid];
    z[b * DK + tid] += sum;
  }
}

// ---------------------------------------------------------------------------
// Per 16-column tile: v = mem@W_mv, gate = sigmoid(mem@W_mb + b),
// prev = (k@W_mem)/den_m, new_info = gate*(v-prev),
// W_mem[:,cols] += k^T @ new_info (K padded 16->32), regen bf16 panels.
// Grid: (64 col-tiles, NB) x 32 threads.
// ---------------------------------------------------------------------------
__global__ __launch_bounds__(32) void kupd_b(const float* __restrict__ memtok,
                                             const ushort_t* __restrict__ panV,
                                             const ushort_t* __restrict__ panB,
                                             ushort_t* __restrict__ panM,
                                             float* __restrict__ Wmem,
                                             const float* __restrict__ kf32,
                                             const ushort_t* __restrict__ kbf,
                                             const float* __restrict__ denm,
                                             const float* __restrict__ bmb) {
  __shared__ float kl[MT * DK];
  __shared__ float ni[MT * 16];
  __shared__ float Wn[DK * 16];
  int nt = blockIdx.x, b = blockIdx.y;
  int l = threadIdx.x, m = l & 15, g = l >> 4;
  for (int i = l; i < MT * DK; i += 32) kl[i] = kf32[(size_t)b * MT * DK + i];
  __syncthreads();
  v8f av = {}, ag = {}, ap = {};
  const float* arow = memtok + ((size_t)b * MT + m) * DIM;
  for (int kt = 0; kt < 32; ++kt) {
    const float* a0 = arow + kt * 32 + 8 * g;
    float t[16];
    *(float4*)(t + 0)  = *(const float4*)(a0);
    *(float4*)(t + 4)  = *(const float4*)(a0 + 4);
    *(float4*)(t + 8)  = *(const float4*)(a0 + 16);
    *(float4*)(t + 12) = *(const float4*)(a0 + 20);
    BF16x16 A;
#pragma unroll
    for (int i = 0; i < 16; ++i) A.u[i] = f2bf(t[i]);
    BF16x16 BvV, BvG;
    const uint4* pv = (const uint4*)(panV + ((size_t)(kt * 64 + nt) * 32 + l) * 16);
    const uint4* pg = (const uint4*)(panB + ((size_t)(kt * 64 + nt) * 32 + l) * 16);
    BvV.q[0] = pv[0]; BvV.q[1] = pv[1];
    BvG.q[0] = pg[0]; BvG.q[1] = pg[1];
    av = wmma_bf16(A.bf, BvV.bf, av);
    ag = wmma_bf16(A.bf, BvG.bf, ag);
  }
  const ushort_t* krow = kbf + ((size_t)b * MT + m) * DK;
  const ushort_t* pmb = panM + (size_t)b * (4 * 64 * 32 * 16);
  for (int kt = 0; kt < 4; ++kt) {
    BF16x16 A;
    const ushort_t* a0 = krow + kt * 32 + 8 * g;
    A.q[0] = *(const uint4*)a0;
    A.q[1] = *(const uint4*)(a0 + 16);
    BF16x16 Bv;
    const uint4* bp = (const uint4*)(pmb + ((size_t)(kt * 64 + nt) * 32 + l) * 16);
    Bv.q[0] = bp[0]; Bv.q[1] = bp[1];
    ap = wmma_bf16(A.bf, Bv.bf, ap);
  }
  int c0 = nt * 16;
#pragma unroll
  for (int r = 0; r < 8; ++r) {
    int t = r + 8 * g;
    float dmn = denm[b * MT + t];
    float gate = 1.f / (1.f + expf(-(ag[r] + bmb[c0 + m])));
    float prev = ap[r] / dmn;
    ni[t * 16 + m] = gate * (av[r] - prev);
  }
  __syncthreads();
  // W_mem[:, cols] += k^T @ new_info  (A: 16 feat x 32 tok-padded, B: 32 x 16)
  for (int mt = 0; mt < 8; ++mt) {
    BF16x16 A, Bv;
#pragma unroll
    for (int i = 0; i < 16; ++i) {
      A.u[i]  = (i < 8) ? f2bf(kl[(8 * g + i) * DK + mt * 16 + m]) : (ushort_t)0;
      Bv.u[i] = (i < 8) ? f2bf(ni[(8 * g + i) * 16 + m]) : (ushort_t)0;
    }
    v8f zc = {};
    v8f dW = wmma_bf16(A.bf, Bv.bf, zc);
#pragma unroll
    for (int r = 0; r < 8; ++r) {
      int f = mt * 16 + r + 8 * g;
      size_t wi = ((size_t)b * DK + f) * DIM + c0 + m;
      float nv = Wmem[wi] + dW[r];
      Wmem[wi] = nv;
      Wn[f * 16 + m] = nv;
    }
  }
  __syncthreads();
  // regenerate bf16 B-panels of W_mem for this column tile
  ushort_t* pout = panM + (size_t)b * (4 * 64 * 32 * 16);
#pragma unroll
  for (int kt = 0; kt < 4; ++kt) {
    ushort_t* o = pout + ((size_t)(kt * 64 + nt) * 32 + l) * 16;
#pragma unroll
    for (int i = 0; i < 16; ++i) {
      int kk = kt * 32 + ((i < 8) ? (8 * g + i) : (16 + 8 * g + (i - 8)));
      o[i] = f2bf(Wn[kk * 16 + m]);
    }
  }
}

// ---------------------------------------------------------------------------
extern "C" void kernel_launch(void* const* d_in, const int* in_sizes, int n_in,
                              void* d_out, int out_size, void* d_ws, size_t ws_size,
                              hipStream_t stream) {
  (void)in_sizes; (void)n_in; (void)out_size; (void)ws_size;
  const float* hid = (const float*)d_in[0];
  const float* mo  = (const float*)d_in[1];
  const float* Wmq = (const float*)d_in[2];
  const float* Wmk = (const float*)d_in[3];
  const float* Wmv = (const float*)d_in[4];
  const float* Wmb = (const float*)d_in[5];
  const float* bmb = (const float*)d_in[6];
  float* out = (float*)d_out;
  char* ws = (char*)d_ws;

  size_t o = 0;
  float*    Wmem = (float*)(ws + o);    o += (size_t)NB * DK * DIM * 4;  // 2 MB
  ushort_t* panM = (ushort_t*)(ws + o); o += (size_t)NB * DK * DIM * 2;  // 1 MB
  float*    zv   = (float*)(ws + o);    o += (size_t)NB * DK * 4;        // 2 KB
  ushort_t* Qbf  = (ushort_t*)(ws + o); o += (size_t)NB * SEQ * DK * 2;  // 4.1 MB
  ushort_t* panQ = (ushort_t*)(ws + o); o += (size_t)DIM * DMQ * 2;
  ushort_t* panK = (ushort_t*)(ws + o); o += (size_t)DIM * DMQ * 2;
  ushort_t* panV = (ushort_t*)(ws + o); o += (size_t)DIM * DIM * 2;      // 2 MB
  ushort_t* panB = (ushort_t*)(ws + o); o += (size_t)DIM * DIM * 2;      // 2 MB
  float*    mtk  = (float*)(ws + o);    o += (size_t)NB * MT * DIM * 4;
  float*    kf   = (float*)(ws + o);    o += (size_t)NB * MT * DK * 4;
  ushort_t* kbf  = (ushort_t*)(ws + o); o += (size_t)NB * MT * DK * 2;
  float*    dnm  = (float*)(ws + o);    o += (size_t)NB * MT * 4;

  // zero [Wmem | panM | zv] (contiguous) -> fresh state each call
  long zwords = ((size_t)NB * DK * DIM * 4 + (size_t)NB * DK * DIM * 2 +
                 (size_t)NB * DK * 4) / 4;
  kzero<<<(unsigned)((zwords + 255) / 256), 256, 0, stream>>>((unsigned*)Wmem, zwords);

  kprep<<<(32 * 4 * 32 + 255) / 256, 256, 0, stream>>>(Wmq, panQ, DIM, DMQ);
  kprep<<<(32 * 4 * 32 + 255) / 256, 256, 0, stream>>>(Wmk, panK, DIM, DMQ);
  kprep<<<(32 * 64 * 32 + 255) / 256, 256, 0, stream>>>(Wmv, panV, DIM, DIM);
  kprep<<<(32 * 64 * 32 + 255) / 256, 256, 0, stream>>>(Wmb, panB, DIM, DIM);

  kq<<<(NB * SEQ) / 16, 32, 0, stream>>>(hid, panQ, Qbf);

  for (int s = 0; s < NSEG; ++s) {
    kassoc<<<dim3(NB * 33, 16), 32, 0, stream>>>(Qbf, panM, zv, mo, out, mtk, s);
    kupd_a<<<NB, 128, 0, stream>>>(mtk, panK, zv, kf, kbf, dnm);
    kupd_b<<<dim3(64, NB), 32, 0, stream>>>(mtk, panV, panB, panM, Wmem, kf, kbf,
                                            dnm, bmb);
  }
}
